// GVPGNN_47098611368286
// MI455X (gfx1250) — matrix operand order repaired and checked by reading, not compiled, not run on hardware
//
#include <hip/hip_runtime.h>
#include <hip/hip_bf16.h>
#include <math.h>

typedef __attribute__((ext_vector_type(16))) _Float16 v16h;
typedef __attribute__((ext_vector_type(8)))  float    v8f;

#define N_NODES_C  10000
#define N_EDGES_C  320000
#define N_GRAPHS_C 20

// ---------------------------------------------------------------------------
// WMMA helpers (CDNA5 / gfx1250, wave32).
// A (16xK f16) staged in LDS row-major; fragment in contiguous k order (this code's, same as B):
//   lane L (row = L&15, half = L>>4), VGPR j holds K pair
//   kofs(j) = 2j + 16*half  ->  dword (k0+kofs)/2
// B (KxN f16) in global "pair-K-major": dword[(k>>1)*N + n] = {W[k][n], W[k+1][n]}
//   lane L (n = L&15, half = L>>4), VGPR j holds K = k0 + 2j + 16*half
// C/D: lane L, reg r -> (M = r + 8*half, N = (L&15) + 16*tile)
// ---------------------------------------------------------------------------
static __device__ __forceinline__ v16h lds_a_frag(const _Float16* A, int row, int half,
                                                  int k0, int stride) {
  union { v16h v; unsigned u[8]; } r;
  const unsigned* p = (const unsigned*)(A + row * stride);
  int base = (k0 >> 1) + 8 * half;
#pragma unroll
  for (int j = 0; j < 8; ++j) r.u[j] = p[base + j];
  return r.v;
}

static __device__ __forceinline__ v16h glb_b_frag(const _Float16* Wp, int N,
                                                  int k0, int n, int half) {
  union { v16h v; unsigned u[8]; } r;
  const unsigned* p = (const unsigned*)Wp;
  int kb = (k0 >> 1) + 8 * half;
#pragma unroll
  for (int j = 0; j < 8; ++j) r.u[j] = p[(kb + j) * N + n];
  return r.v;
}

static __device__ __forceinline__ v8f wmma_f16(v16h a, v16h b, v8f c) {
  return __builtin_amdgcn_wmma_f32_16x16x32_f16(false, a, false, b, (short)0, c,
                                                false, false);
}

// 16xK (LDS) @ Kx128 (global prepped) -> acc[8] in C layout
static __device__ __forceinline__ void gemm_16xKx128(const _Float16* A, int stride,
                                                     const _Float16* Wp, int KC,
                                                     int L, v8f acc[8]) {
  const int row = L & 15, half = L >> 4, nb = L & 15;
  for (int kc = 0; kc < KC; ++kc) {
    v16h a = lds_a_frag(A, row, half, kc * 32, stride);
    if (kc + 1 < KC) __builtin_prefetch(Wp + ((kc + 1) * 16) * 128, 0, 1);
#pragma unroll
    for (int t = 0; t < 8; ++t) {
      v16h b = glb_b_frag(Wp, 128, kc * 32, t * 16 + nb, half);
      acc[t] = wmma_f16(a, b, acc[t]);
    }
  }
}

// 16xK (LDS) @ Kx16 (global prepped) -> single C-tile (gate matmuls)
static __device__ __forceinline__ v8f gemm_16xKx16(const _Float16* A, int stride,
                                                   const _Float16* Gp, int KC, int L) {
  v8f acc = {};
  const int row = L & 15, half = L >> 4, nb = L & 15;
  for (int kc = 0; kc < KC; ++kc) {
    v16h a = lds_a_frag(A, row, half, kc * 32, stride);
    v16h b = glb_b_frag(Gp, 16, kc * 32, nb, half);
    acc = wmma_f16(a, b, acc);
  }
  return acc;
}

static __device__ __forceinline__ float wave_sum(float x) {
#pragma unroll
  for (int m = 16; m; m >>= 1) x += __shfl_xor(x, m, 32);
  return x;
}
static __device__ __forceinline__ float sigmoidf_(float x) {
  return 1.f / (1.f + __expf(-x));
}

// ---------------------------------------------------------------------------
// Utility kernels
// ---------------------------------------------------------------------------
__global__ void zero_kernel(float* __restrict__ p, int n) {
  int i = blockIdx.x * blockDim.x + threadIdx.x;
  if (i < n) p[i] = 0.f;
}

// f32 (KxN) -> f16 pair-K-major with zero padding to Kpad
__global__ void prep_w_kernel(const float* __restrict__ W, _Float16* __restrict__ Wp,
                              int K, int Kpad, int N) {
  int i = blockIdx.x * blockDim.x + threadIdx.x;
  if (i >= Kpad * N) return;
  int k = i / N, n = i % N;
  float v = (k < K) ? W[k * N + n] : 0.f;
  Wp[((k >> 1) * N + n) * 2 + (k & 1)] = (_Float16)v;
}

// out = x + dx / clip(deg,1)
__global__ void agg_kernel(const float* __restrict__ x, const float* __restrict__ dx,
                           const float* __restrict__ deg, float* __restrict__ out,
                           int cols, int total) {
  int i = blockIdx.x * blockDim.x + threadIdx.x;
  if (i >= total) return;
  float d = fmaxf(deg[i / cols], 1.f);
  out[i] = x[i] + dx[i] / d;
}

// LayerNorm on s (128) + vector-norm rescale of v (16x3); one wave per node
__global__ __launch_bounds__(256)
void ln_kernel(const float* __restrict__ s_in, const float* __restrict__ v_in,
               const float* __restrict__ g, const float* __restrict__ b,
               float* __restrict__ s_out, float* __restrict__ v_out) {
  const int wid = threadIdx.x >> 5, L = threadIdx.x & 31;
  const int nd = blockIdx.x * 8 + wid;
  float x[4]; float sum = 0.f;
#pragma unroll
  for (int k = 0; k < 4; ++k) { x[k] = s_in[nd * 128 + L + 32 * k]; sum += x[k]; }
  float mu = wave_sum(sum) / 128.f;
  float vs = 0.f;
#pragma unroll
  for (int k = 0; k < 4; ++k) { float d = x[k] - mu; vs += d * d; }
  float inv = rsqrtf(wave_sum(vs) / 128.f + 1e-5f);
#pragma unroll
  for (int k = 0; k < 4; ++k) {
    int n = L + 32 * k;
    s_out[nd * 128 + n] = (x[k] - mu) * inv * g[n] + b[n];
  }
  float vn = 0.f;
  if (L < 16) {
    float a = v_in[nd * 48 + L * 3], c = v_in[nd * 48 + L * 3 + 1],
          d = v_in[nd * 48 + L * 3 + 2];
    vn = fmaxf(a * a + c * c + d * d, 1e-8f);
  }
  float scale = rsqrtf(wave_sum(vn) / 16.f);
  for (int t = L; t < 48; t += 32) v_out[nd * 48 + t] = v_in[nd * 48 + t] * scale;
}

// h = input@emb_w ; LN ; s0 = h@ws_w + b  (one wave per node)
__global__ __launch_bounds__(256)
void embed_kernel(const float* __restrict__ inp, const float* __restrict__ embW,
                  const float* __restrict__ lg, const float* __restrict__ lb,
                  const float* __restrict__ wsw, const float* __restrict__ wsb,
                  float* __restrict__ s_out) {
  __shared__ float hl_sh[8][24];
  const int wid = threadIdx.x >> 5, L = threadIdx.x & 31;
  const int nd = blockIdx.x * 8 + wid;
  float h = 0.f;
  if (L < 21) for (int k = 0; k < 21; ++k) h += inp[nd * 21 + k] * embW[k * 21 + L];
  float mu = wave_sum(L < 21 ? h : 0.f) / 21.f;
  float d = (L < 21) ? (h - mu) : 0.f;
  float inv = rsqrtf(wave_sum(d * d) / 21.f + 1e-5f);
  if (L < 21) hl_sh[wid][L] = (h - mu) * inv * lg[L] + lb[L];
  __syncthreads();
  const float* hl = hl_sh[wid];
#pragma unroll
  for (int k = 0; k < 4; ++k) {
    int n = L + 32 * k;
    float a = wsb[n];
    for (int j = 0; j < 21; ++j) a += hl[j] * wsw[j * 128 + n];
    s_out[nd * 128 + n] = a;
  }
}

// edge RBF + LN + we_gvp  (one wave per edge) -> es_h (f16), ev (f32)
__global__ __launch_bounds__(256)
void edge_feat_kernel(const float* __restrict__ pos, const int* __restrict__ src,
                      const int* __restrict__ dst, const float* __restrict__ lg,
                      const float* __restrict__ lb, const float* __restrict__ wh,
                      const float* __restrict__ wsw, const float* __restrict__ wvv,
                      const float* __restrict__ wsvw, const float* __restrict__ wsvb,
                      const float* __restrict__ wsb, _Float16* __restrict__ es_h,
                      float* __restrict__ ev) {
  __shared__ float cat_sh[8][34];
  const int wid = threadIdx.x >> 5, L = threadIdx.x & 31;
  const int e = blockIdx.x * 8 + wid;
  int si = src[e], di = dst[e];
  float vx = pos[di * 3] - pos[si * 3];
  float vy = pos[di * 3 + 1] - pos[si * 3 + 1];
  float vz = pos[di * 3 + 2] - pos[si * 3 + 2];
  float d2 = vx * vx + vy * vy + vz * vz;
  float dd = sqrtf(d2 + 1e-12f);
  float muj = (float)L * (20.f / 31.f);
  float t0 = (dd - muj) / 0.625f;
  float es = __expf(-t0 * t0);
  float mu = wave_sum(es) / 32.f;
  float dv_ = es - mu;
  float inv = rsqrtf(wave_sum(dv_ * dv_) / 32.f + 1e-5f);
  float esl = (es - mu) * inv * lg[L] + lb[L];
  float scale = rsqrtf(fmaxf(d2, 1e-8f));
  float w00 = wh[0];
  float vhx = vx * scale * w00, vhy = vy * scale * w00, vhz = vz * scale * w00;
  float vnh = sqrtf(fmaxf(vhx * vhx + vhy * vhy + vhz * vhz, 1e-8f));
  cat_sh[wid][L] = esl;
  if (L == 0) cat_sh[wid][32] = vnh;
  __syncthreads();
  const float* cat = cat_sh[wid];
  float o = wsb[L];
  for (int j = 0; j < 33; ++j) o += cat[j] * wsw[j * 32 + L];
  float gate = sigmoidf_(wave_sum(o * wsvw[L]) + wsvb[0]);
  es_h[e * 32 + L] = (_Float16)o;
  if (L == 0) {
    float wv0 = wvv[0];
    ev[e * 3 + 0] = vhx * wv0 * gate;
    ev[e * 3 + 1] = vhy * wv0 * gate;
    ev[e * 3 + 2] = vhz * wv0 * gate;
  }
}

// ---------------------------------------------------------------------------
// Fused per-edge conv layer: GVP0(321->128) -> GVP1/2(144->128) + gates + vouts,
// scatter-add to nodes.  Block = 64 threads = 2 waves, 16 edges per wave.
// ---------------------------------------------------------------------------
struct ConvW {
  const float *wh0, *wv0, *svb0, *b0;
  const float *wh1, *wv1, *svb1, *b1;
  const float *wh2, *wv2, *svb2, *b2;
  const _Float16 *W0p, *W1p, *W2p, *G0p, *G1p, *G2p;
};

// GVP stage with 16 vector channels (conv[1]/conv[2]); A cols 144..159 must be 0.
static __device__ __forceinline__ void conv_gvp16(_Float16* A, float* MV, float* VH,
                                                  const float* wh, const _Float16* Wp,
                                                  const float* b, const _Float16* Gp,
                                                  const float* svb, const float* wv,
                                                  bool do_relu, int L, v8f acc[8]) {
  const int half = L >> 4, nb = L & 15;
  for (int t = L; t < 768; t += 32) {  // vh[e][h][c] = sum_v MV * wh
    int e = t / 48, r = t % 48, h = r / 3, c = r % 3;
    float a = 0.f;
    for (int vv = 0; vv < 16; ++vv) a += MV[e * 48 + vv * 3 + c] * wh[vv * 16 + h];
    VH[e * 48 + h * 3 + c] = a;
  }
  __syncthreads();
  for (int t = L; t < 256; t += 32) {  // vn -> A cols 128..143
    int e = t >> 4, h = t & 15;
    float x = VH[e * 48 + h * 3], y = VH[e * 48 + h * 3 + 1], z = VH[e * 48 + h * 3 + 2];
    A[e * 352 + 128 + h] = (_Float16)sqrtf(fmaxf(x * x + y * y + z * z, 1e-8f));
  }
  __syncthreads();
  v8f zero = {};
#pragma unroll
  for (int t = 0; t < 8; ++t) acc[t] = zero;
  gemm_16xKx128(A, 352, Wp, 5, L, acc);
#pragma unroll
  for (int t = 0; t < 8; ++t) {
    float bb = b[t * 16 + nb];
#pragma unroll
    for (int r = 0; r < 8; ++r) acc[t][r] += bb;
  }
  // stage pre-activation s for the gate GEMM (gate uses pre-act per reference)
#pragma unroll
  for (int t = 0; t < 8; ++t)
#pragma unroll
    for (int r = 0; r < 8; ++r)
      A[(r + 8 * half) * 352 + t * 16 + nb] = (_Float16)acc[t][r];
  __syncthreads();
  v8f ga = gemm_16xKx16(A, 352, Gp, 4, L);
  float gt[8];
  float sb = svb[nb];
#pragma unroll
  for (int r = 0; r < 8; ++r) gt[r] = sigmoidf_(ga[r] + sb);
  if (do_relu) {
#pragma unroll
    for (int t = 0; t < 8; ++t)
#pragma unroll
      for (int r = 0; r < 8; ++r)
        A[(r + 8 * half) * 352 + t * 16 + nb] = (_Float16)fmaxf(acc[t][r], 0.f);
  }
  __syncthreads();
#pragma unroll
  for (int r = 0; r < 8; ++r) {  // mv_out[e][v=nb][c] = gate * (vh . wv)
    int e = r + 8 * half;
#pragma unroll
    for (int c = 0; c < 3; ++c) {
      float a = 0.f;
      for (int h = 0; h < 16; ++h) a += VH[e * 48 + h * 3 + c] * wv[h * 16 + nb];
      MV[e * 48 + nb * 3 + c] = a * gt[r];
    }
  }
  __syncthreads();
}

__global__ __launch_bounds__(64)
void edge_msg_kernel(ConvW cw, const float* __restrict__ s, const float* __restrict__ v,
                     const _Float16* __restrict__ es_h, const float* __restrict__ ev,
                     const int* __restrict__ src, const int* __restrict__ dst,
                     float* __restrict__ ds, float* __restrict__ dv,
                     float* __restrict__ deg) {
  __shared__ _Float16 A_sh[2][16 * 352];
  __shared__ float MV_sh[2][16 * 99];
  __shared__ float VH_sh[2][16 * 99];
  __shared__ int IDX_sh[2][32];
  const int wid = threadIdx.x >> 5, L = threadIdx.x & 31;
  const int half = L >> 4, nb = L & 15;
  _Float16* A = A_sh[wid];
  float* MV = MV_sh[wid];
  float* VH = VH_sh[wid];
  int* IDX = IDX_sh[wid];
  const int e0 = blockIdx.x * 32 + wid * 16;

  IDX[L] = (L < 16) ? src[e0 + L] : dst[e0 + (L - 16)];
  __syncthreads();

  // gather mv = [v[src](16) | ev(1) | v[dst](16)] x 3
  for (int t = L; t < 16 * 99; t += 32) {
    int e = t / 99, r = t % 99, vv = r / 3, c = r % 3;
    float val;
    if (vv < 16)       val = v[IDX[e] * 48 + vv * 3 + c];
    else if (vv == 16) val = ev[(e0 + e) * 3 + c];
    else               val = v[IDX[16 + e] * 48 + (vv - 17) * 3 + c];
    MV[t] = val;
  }
  // scalars -> A : [s_src 0..127 | es 128..159 | s_dst 160..287 | vn 288..320 | pad]
  for (int t = L; t < 16 * 128; t += 32) {
    int e = t >> 7, n = t & 127;
    A[e * 352 + n]       = (_Float16)s[IDX[e] * 128 + n];
    A[e * 352 + 160 + n] = (_Float16)s[IDX[16 + e] * 128 + n];
  }
  for (int t = L; t < 16 * 32; t += 32) {
    int e = t >> 5, n = t & 31;
    A[e * 352 + 128 + n] = es_h[(e0 + e) * 32 + n];
    A[e * 352 + 320 + n] = (_Float16)0.f;  // pad 320..351 (col 320 rewritten by vn)
  }
  __syncthreads();
  // vh0[e][h<33][c]
  for (int t = L; t < 16 * 99; t += 32) {
    int e = t / 99, r = t % 99, h = r / 3, c = r % 3;
    float a = 0.f;
    for (int vv = 0; vv < 33; ++vv) a += MV[e * 99 + vv * 3 + c] * cw.wh0[vv * 33 + h];
    VH[e * 99 + h * 3 + c] = a;
  }
  __syncthreads();
  for (int t = L; t < 16 * 33; t += 32) {
    int e = t / 33, h = t % 33;
    float x = VH[e * 99 + h * 3], y = VH[e * 99 + h * 3 + 1], z = VH[e * 99 + h * 3 + 2];
    A[e * 352 + 288 + h] = (_Float16)sqrtf(fmaxf(x * x + y * y + z * z, 1e-8f));
  }
  __syncthreads();

  // ---- GVP0: (16x352) @ (352x128) via WMMA ----
  v8f acc[8];
  v8f zero = {};
#pragma unroll
  for (int t = 0; t < 8; ++t) acc[t] = zero;
  gemm_16xKx128(A, 352, cw.W0p, 11, L, acc);
#pragma unroll
  for (int t = 0; t < 8; ++t) {
    float bb = cw.b0[t * 16 + nb];
#pragma unroll
    for (int r = 0; r < 8; ++r) acc[t][r] += bb;
  }
#pragma unroll
  for (int t = 0; t < 8; ++t)  // pre-act s1 -> A cols 0..127
#pragma unroll
    for (int r = 0; r < 8; ++r)
      A[(r + 8 * half) * 352 + t * 16 + nb] = (_Float16)acc[t][r];
  for (int t = L; t < 256; t += 32) {  // zero pad cols 144..159 for GVP1/2 GEMMs
    int e = t >> 4, n = t & 15;
    A[e * 352 + 144 + n] = (_Float16)0.f;
  }
  __syncthreads();
  v8f ga = gemm_16xKx16(A, 352, cw.G0p, 4, L);
  float gt[8];
  {
    float sb = cw.svb0[nb];
#pragma unroll
    for (int r = 0; r < 8; ++r) gt[r] = sigmoidf_(ga[r] + sb);
  }
#pragma unroll
  for (int t = 0; t < 8; ++t)  // relu(s1) -> A cols 0..127
#pragma unroll
    for (int r = 0; r < 8; ++r)
      A[(r + 8 * half) * 352 + t * 16 + nb] = (_Float16)fmaxf(acc[t][r], 0.f);
#pragma unroll
  for (int r = 0; r < 8; ++r) {  // mv1[e][v=nb][c]
    int e = r + 8 * half;
#pragma unroll
    for (int c = 0; c < 3; ++c) {
      float a = 0.f;
      for (int h = 0; h < 33; ++h) a += VH[e * 99 + h * 3 + c] * cw.wv0[h * 16 + nb];
      MV[e * 48 + nb * 3 + c] = a * gt[r];
    }
  }
  __syncthreads();

  conv_gvp16(A, MV, VH, cw.wh1, cw.W1p, cw.b1, cw.G1p, cw.svb1, cw.wv1, true, L, acc);
  conv_gvp16(A, MV, VH, cw.wh2, cw.W2p, cw.b2, cw.G2p, cw.svb2, cw.wv2, false, L, acc);

  // scatter-add messages to dst nodes
#pragma unroll
  for (int t = 0; t < 8; ++t)
#pragma unroll
    for (int r = 0; r < 8; ++r) {
      int e = r + 8 * half;
      atomicAdd(&ds[IDX[16 + e] * 128 + t * 16 + nb], acc[t][r]);
    }
  for (int t = L; t < 16 * 48; t += 32) {
    int e = t / 48, r = t % 48;
    atomicAdd(&dv[IDX[16 + e] * 48 + r], MV[t]);
  }
  if (L < 16) atomicAdd(&deg[IDX[16 + L]], 1.0f);
}

// ---------------------------------------------------------------------------
// Fused node feed-forward: GVP(160->512, relu) + GVP(544->128) with residuals.
// One wave per block, 16 nodes per wave.
// ---------------------------------------------------------------------------
struct FFW {
  const float *whf0, *wvf0, *svbf0, *bf0;
  const float *whf1, *wvf1, *svbf1, *bf1;
  const _Float16 *Wf0p, *Gf0p, *Wf1p, *Gf1p;
};

__global__ __launch_bounds__(32)
void ff_kernel(FFW fw, const float* __restrict__ s_mid, const float* __restrict__ v_mid,
               float* __restrict__ s_tmp, float* __restrict__ v_tmp) {
  __shared__ _Float16 A0[16 * 160];
  __shared__ _Float16 AB[16 * 544];
  __shared__ float VH[16 * 96];
  __shared__ float MV[16 * 96];
  const int L = threadIdx.x;
  const int row = L & 15, half = L >> 4, nb = L & 15;
  const int n0 = blockIdx.x * 16;

  for (int t = L; t < 16 * 128; t += 32) {
    int e = t >> 7, n = t & 127;
    A0[e * 160 + n] = (_Float16)s_mid[(n0 + e) * 128 + n];
  }
  for (int t = L; t < 16 * 96; t += 32) {  // vh0[e][h<32][c]
    int e = t / 96, r = t % 96, h = r / 3, c = r % 3;
    float a = 0.f;
    for (int vv = 0; vv < 16; ++vv)
      a += v_mid[(n0 + e) * 48 + vv * 3 + c] * fw.whf0[vv * 32 + h];
    VH[t] = a;
  }
  __syncthreads();
  for (int t = L; t < 16 * 32; t += 32) {
    int e = t >> 5, h = t & 31;
    float x = VH[e * 96 + h * 3], y = VH[e * 96 + h * 3 + 1], z = VH[e * 96 + h * 3 + 2];
    A0[e * 160 + 128 + h] = (_Float16)sqrtf(fmaxf(x * x + y * y + z * z, 1e-8f));
  }
  __syncthreads();
  v8f zero = {};
  for (int g = 0; g < 4; ++g) {  // (16x160)@(160x512) in 4 column groups
    v8f acc[8];
#pragma unroll
    for (int t = 0; t < 8; ++t) acc[t] = zero;
    for (int kc = 0; kc < 5; ++kc) {
      v16h a = lds_a_frag(A0, row, half, kc * 32, 160);
#pragma unroll
      for (int t = 0; t < 8; ++t) {
        v16h b = glb_b_frag(fw.Wf0p, 512, kc * 32, g * 128 + t * 16 + nb, half);
        acc[t] = wmma_f16(a, b, acc[t]);
      }
    }
#pragma unroll
    for (int t = 0; t < 8; ++t) {
      int n = g * 128 + t * 16 + nb;
      float bb = fw.bf0[n];
#pragma unroll
      for (int r = 0; r < 8; ++r)
        AB[(r + 8 * half) * 544 + n] = (_Float16)(acc[t][r] + bb);
    }
  }
  __syncthreads();
  v8f g0a = zero, g0b = zero;  // gate0 from pre-act (K=512, N=32)
  for (int kc = 0; kc < 16; ++kc) {
    v16h a = lds_a_frag(AB, row, half, kc * 32, 544);
    v16h b0 = glb_b_frag(fw.Gf0p, 32, kc * 32, nb, half);
    v16h b1 = glb_b_frag(fw.Gf0p, 32, kc * 32, 16 + nb, half);
    g0a = wmma_f16(a, b0, g0a);
    g0b = wmma_f16(a, b1, g0b);
  }
  float gtA[8], gtB[8];
  {
    float s0 = fw.svbf0[nb], s1 = fw.svbf0[16 + nb];
#pragma unroll
    for (int r = 0; r < 8; ++r) {
      gtA[r] = sigmoidf_(g0a[r] + s0);
      gtB[r] = sigmoidf_(g0b[r] + s1);
    }
  }
  for (int t = L; t < 16 * 512; t += 32) {  // relu in place
    int e = t >> 9, n = t & 511;
    float x = (float)AB[e * 544 + n];
    AB[e * 544 + n] = (_Float16)fmaxf(x, 0.f);
  }
  __syncthreads();
#pragma unroll
  for (int r = 0; r < 8; ++r) {  // mv_f[e][v<32][c]
    int e = r + 8 * half;
#pragma unroll
    for (int c = 0; c < 3; ++c) {
      float aA = 0.f, aB = 0.f;
      for (int h = 0; h < 32; ++h) {
        float vh = VH[e * 96 + h * 3 + c];
        aA += vh * fw.wvf0[h * 32 + nb];
        aB += vh * fw.wvf0[h * 32 + 16 + nb];
      }
      MV[e * 96 + nb * 3 + c] = aA * gtA[r];
      MV[e * 96 + (16 + nb) * 3 + c] = aB * gtB[r];
    }
  }
  __syncthreads();
  for (int t = L; t < 16 * 96; t += 32) {  // vh1[e][h<32][c]
    int e = t / 96, r = t % 96, h = r / 3, c = r % 3;
    float a = 0.f;
    for (int vv = 0; vv < 32; ++vv) a += MV[e * 96 + vv * 3 + c] * fw.whf1[vv * 32 + h];
    VH[t] = a;
  }
  __syncthreads();
  for (int t = L; t < 16 * 32; t += 32) {
    int e = t >> 5, h = t & 31;
    float x = VH[e * 96 + h * 3], y = VH[e * 96 + h * 3 + 1], z = VH[e * 96 + h * 3 + 2];
    AB[e * 544 + 512 + h] = (_Float16)sqrtf(fmaxf(x * x + y * y + z * z, 1e-8f));
  }
  __syncthreads();
  v8f acc[8];  // (16x544)@(544x128)
#pragma unroll
  for (int t = 0; t < 8; ++t) acc[t] = zero;
  for (int kc = 0; kc < 17; ++kc) {
    v16h a = lds_a_frag(AB, row, half, kc * 32, 544);
#pragma unroll
    for (int t = 0; t < 8; ++t) {
      v16h b = glb_b_frag(fw.Wf1p, 128, kc * 32, t * 16 + nb, half);
      acc[t] = wmma_f16(a, b, acc[t]);
    }
  }
#pragma unroll
  for (int t = 0; t < 8; ++t) {
    int n = t * 16 + nb;
    float bb = fw.bf1[n];
#pragma unroll
    for (int r = 0; r < 8; ++r) {
      float s2 = acc[t][r] + bb;
      int e = r + 8 * half;
      s_tmp[(n0 + e) * 128 + n] = s_mid[(n0 + e) * 128 + n] + s2;  // residual
      A0[e * 160 + n] = (_Float16)s2;                              // stage for gate1
    }
  }
  __syncthreads();
  v8f g1 = gemm_16xKx16(A0, 160, fw.Gf1p, 4, L);
  float sb = fw.svbf1[nb];
#pragma unroll
  for (int r = 0; r < 8; ++r) {
    int e = r + 8 * half;
    float gate = sigmoidf_(g1[r] + sb);
#pragma unroll
    for (int c = 0; c < 3; ++c) {
      float a = 0.f;
      for (int h = 0; h < 32; ++h) a += VH[e * 96 + h * 3 + c] * fw.wvf1[h * 16 + nb];
      v_tmp[(n0 + e) * 48 + nb * 3 + c] =
          v_mid[(n0 + e) * 48 + nb * 3 + c] + a * gate;  // residual
    }
  }
}

// ---------------------------------------------------------------------------
// Output head: LN + wout GVP + relu; write node features + graph pooling
// ---------------------------------------------------------------------------
__global__ __launch_bounds__(256)
void output_kernel(const float* __restrict__ s, const float* __restrict__ v,
                   const float* __restrict__ lg, const float* __restrict__ lb,
                   const float* __restrict__ wh, const float* __restrict__ wsw,
                   const float* __restrict__ wsb, const int* __restrict__ n2g,
                   float* __restrict__ out) {
  __shared__ float cat_sh[8][144];
  const int wid = threadIdx.x >> 5, L = threadIdx.x & 31;
  const int nd = blockIdx.x * 8 + wid;
  float* cat = cat_sh[wid];
  float x[4]; float sum = 0.f;
#pragma unroll
  for (int k = 0; k < 4; ++k) { x[k] = s[nd * 128 + L + 32 * k]; sum += x[k]; }
  float mu = wave_sum(sum) / 128.f;
  float vs = 0.f;
#pragma unroll
  for (int k = 0; k < 4; ++k) { float d = x[k] - mu; vs += d * d; }
  float inv = rsqrtf(wave_sum(vs) / 128.f + 1e-5f);
#pragma unroll
  for (int k = 0; k < 4; ++k) {
    int n = L + 32 * k;
    cat[n] = (x[k] - mu) * inv * lg[n] + lb[n];
  }
  float vn = 0.f;
  if (L < 16) {
    float a = v[nd * 48 + L * 3], c = v[nd * 48 + L * 3 + 1], d = v[nd * 48 + L * 3 + 2];
    vn = fmaxf(a * a + c * c + d * d, 1e-8f);
  }
  float scale = rsqrtf(wave_sum(vn) / 16.f);
  if (L < 16) {
    float ax = 0.f, ay = 0.f, az = 0.f;
    for (int vv = 0; vv < 16; ++vv) {
      float w = wh[vv * 16 + L];
      ax += v[nd * 48 + vv * 3] * scale * w;
      ay += v[nd * 48 + vv * 3 + 1] * scale * w;
      az += v[nd * 48 + vv * 3 + 2] * scale * w;
    }
    cat[128 + L] = sqrtf(fmaxf(ax * ax + ay * ay + az * az, 1e-8f));
  }
  __syncthreads();
  int gid = n2g[nd];
#pragma unroll
  for (int k = 0; k < 4; ++k) {
    int n = L + 32 * k;
    float a = wsb[n];
    for (int j = 0; j < 144; ++j) a += cat[j] * wsw[j * 128 + n];
    a = fmaxf(a, 0.f);
    out[N_GRAPHS_C * 128 + nd * 128 + n] = a;  // node_feature
    atomicAdd(&out[gid * 128 + n], a);         // graph_feature
  }
}

// ---------------------------------------------------------------------------
// Host side
// ---------------------------------------------------------------------------
static void zero_f(float* p, int n, hipStream_t st) {
  zero_kernel<<<(n + 255) / 256, 256, 0, st>>>(p, n);
}
static void prep_w(const float* W, _Float16* Wp, int K, int Kpad, int N, hipStream_t st) {
  int total = Kpad * N;
  prep_w_kernel<<<(total + 255) / 256, 256, 0, st>>>(W, Wp, K, Kpad, N);
}

extern "C" void kernel_launch(void* const* d_in, const int* in_sizes, int n_in,
                              void* d_out, int out_size, void* d_ws, size_t ws_size,
                              hipStream_t stream) {
  (void)in_sizes; (void)n_in; (void)out_size; (void)ws_size;
  const float* input = (const float*)d_in[0];
  const float* pos   = (const float*)d_in[1];
  const int*   eidx  = (const int*)d_in[2];
  const int*   n2g   = (const int*)d_in[3];
  const int* src = eidx;
  const int* dst = eidx + N_EDGES_C;
  auto f = [&](int i) -> const float* { return (const float*)d_in[4 + i]; };
  // param leaf indices (insertion-order DFS of the params dict):
  // 0 emb_w | 1,2 wv_ln g,b | 3,4 wv_gvp ws_w,ws_b | 5,6 we_ln g,b
  // 7..12 we_gvp wh,ws_w,wv,wsv_w,wsv_b,ws_b | 13+34*l layers | 115.. wout
  char* w = (char*)d_ws;
  auto alloc = [&](size_t bytes) -> void* {
    void* p = (void*)w;
    w += (bytes + 255) & ~(size_t)255;
    return p;
  };
  float* s     = (float*)alloc((size_t)N_NODES_C * 128 * 4);
  float* v     = (float*)alloc((size_t)N_NODES_C * 48 * 4);
  float* s_mid = (float*)alloc((size_t)N_NODES_C * 128 * 4);
  float* v_mid = (float*)alloc((size_t)N_NODES_C * 48 * 4);
  float* s_tmp = (float*)alloc((size_t)N_NODES_C * 128 * 4);
  float* v_tmp = (float*)alloc((size_t)N_NODES_C * 48 * 4);
  float* ds    = (float*)alloc((size_t)N_NODES_C * 128 * 4);
  float* dv    = (float*)alloc((size_t)N_NODES_C * 48 * 4);
  float* deg   = (float*)alloc((size_t)N_NODES_C * 4);
  _Float16* es_h = (_Float16*)alloc((size_t)N_EDGES_C * 32 * 2);
  float* ev    = (float*)alloc((size_t)N_EDGES_C * 3 * 4);
  _Float16 *W0p[3], *W1p[3], *W2p[3], *G0p[3], *G1p[3], *G2p[3];
  _Float16 *Wf0p[3], *Gf0p[3], *Wf1p[3], *Gf1p[3];
  for (int l = 0; l < 3; ++l) {
    W0p[l]  = (_Float16*)alloc(352 * 128 * 2);
    W1p[l]  = (_Float16*)alloc(160 * 128 * 2);
    W2p[l]  = (_Float16*)alloc(160 * 128 * 2);
    G0p[l]  = (_Float16*)alloc(128 * 16 * 2);
    G1p[l]  = (_Float16*)alloc(128 * 16 * 2);
    G2p[l]  = (_Float16*)alloc(128 * 16 * 2);
    Wf0p[l] = (_Float16*)alloc(160 * 512 * 2);
    Gf0p[l] = (_Float16*)alloc(512 * 32 * 2);
    Wf1p[l] = (_Float16*)alloc(544 * 128 * 2);
    Gf1p[l] = (_Float16*)alloc(128 * 16 * 2);
  }
  // prep weights (f32 -> f16 pair-K-major, zero padded)
  for (int l = 0; l < 3; ++l) {
    int B = 13 + 34 * l;
    prep_w(f(B + 1),  W0p[l], 321, 352, 128, stream);
    prep_w(f(B + 7),  W1p[l], 144, 160, 128, stream);
    prep_w(f(B + 13), W2p[l], 144, 160, 128, stream);
    prep_w(f(B + 3),  G0p[l], 128, 128, 16,  stream);
    prep_w(f(B + 9),  G1p[l], 128, 128, 16,  stream);
    prep_w(f(B + 15), G2p[l], 128, 128, 16,  stream);
    prep_w(f(B + 23), Wf0p[l], 160, 160, 512, stream);
    prep_w(f(B + 25), Gf0p[l], 512, 512, 32,  stream);
    prep_w(f(B + 29), Wf1p[l], 544, 544, 128, stream);
    prep_w(f(B + 31), Gf1p[l], 128, 128, 16,  stream);
  }
  // init
  zero_f(v, N_NODES_C * 48, stream);
  zero_f((float*)d_out, N_GRAPHS_C * 128, stream);
  // embedding + edge features
  embed_kernel<<<N_NODES_C / 8, 256, 0, stream>>>(input, f(0), f(1), f(2), f(3), f(4), s);
  edge_feat_kernel<<<N_EDGES_C / 8, 256, 0, stream>>>(pos, src, dst, f(5), f(6), f(7),
                                                      f(8), f(9), f(10), f(11), f(12),
                                                      es_h, ev);
  // conv layers
  for (int l = 0; l < 3; ++l) {
    int B = 13 + 34 * l;
    ConvW cw;
    cw.wh0 = f(B + 0);  cw.wv0 = f(B + 2);  cw.svb0 = f(B + 4);  cw.b0 = f(B + 5);
    cw.wh1 = f(B + 6);  cw.wv1 = f(B + 8);  cw.svb1 = f(B + 10); cw.b1 = f(B + 11);
    cw.wh2 = f(B + 12); cw.wv2 = f(B + 14); cw.svb2 = f(B + 16); cw.b2 = f(B + 17);
    cw.W0p = W0p[l]; cw.W1p = W1p[l]; cw.W2p = W2p[l];
    cw.G0p = G0p[l]; cw.G1p = G1p[l]; cw.G2p = G2p[l];
    FFW fw;
    fw.whf0 = f(B + 22); fw.wvf0 = f(B + 24); fw.svbf0 = f(B + 26); fw.bf0 = f(B + 27);
    fw.whf1 = f(B + 28); fw.wvf1 = f(B + 30); fw.svbf1 = f(B + 32); fw.bf1 = f(B + 33);
    fw.Wf0p = Wf0p[l]; fw.Gf0p = Gf0p[l]; fw.Wf1p = Wf1p[l]; fw.Gf1p = Gf1p[l];

    zero_f(ds, N_NODES_C * 128, stream);
    zero_f(dv, N_NODES_C * 48, stream);
    zero_f(deg, N_NODES_C, stream);
    edge_msg_kernel<<<N_EDGES_C / 32, 64, 0, stream>>>(cw, s, v, es_h, ev, src, dst,
                                                       ds, dv, deg);
    agg_kernel<<<(N_NODES_C * 128 + 255) / 256, 256, 0, stream>>>(s, ds, deg, s_tmp,
                                                                  128, N_NODES_C * 128);
    agg_kernel<<<(N_NODES_C * 48 + 255) / 256, 256, 0, stream>>>(v, dv, deg, v_tmp,
                                                                 48, N_NODES_C * 48);
    ln_kernel<<<N_NODES_C / 8, 256, 0, stream>>>(s_tmp, v_tmp, f(B + 18), f(B + 19),
                                                 s_mid, v_mid);
    ff_kernel<<<N_NODES_C / 16, 32, 0, stream>>>(fw, s_mid, v_mid, s_tmp, v_tmp);
    ln_kernel<<<N_NODES_C / 8, 256, 0, stream>>>(s_tmp, v_tmp, f(B + 20), f(B + 21),
                                                 s, v);
  }
  // output head
  output_kernel<<<N_NODES_C / 8, 256, 0, stream>>>(s, v, f(115), f(116), f(117),
                                                   f(118), f(119), n2g, (float*)d_out);
}
